// EdgeAugmentedGraphTransformer_30013231464553
// MI455X (gfx1250) — compile-verified, matrix operand order checked by reading, not run on hardware
//
#include <hip/hip_runtime.h>
#include <hip/hip_bf16.h>

typedef __attribute__((ext_vector_type(16))) _Float16 v16h;
typedef __attribute__((ext_vector_type(8)))  _Float16 h8;
typedef __attribute__((ext_vector_type(8)))  float    v8f;

#define HIDD  512
#define HEADS 8
#define CH    64
#define EDIM  32

// ---------------------------------------------------------------------------
// float atomic max via signed/unsigned int trick (init must be -inf)
// ---------------------------------------------------------------------------
__device__ __forceinline__ void atomicMaxF(float* addr, float val) {
  if (val >= 0.0f)
    atomicMax((int*)addr, __float_as_int(val));
  else
    atomicMin((unsigned int*)addr, __float_as_uint(val));
}

// ---------------------------------------------------------------------------
// f32 -> f16 convert (plain)
// ---------------------------------------------------------------------------
__global__ void cvt_kernel(const float* __restrict__ src, _Float16* __restrict__ dst, int n) {
  int i = blockIdx.x * blockDim.x + threadIdx.x;
  if (i < n) dst[i] = (_Float16)src[i];
}

// ---------------------------------------------------------------------------
// f32 [K,N] -> f16 [N,K] transpose-convert (weights; amortized over all tiles)
// ---------------------------------------------------------------------------
__global__ void transpose_cvt_kernel(const float* __restrict__ src, _Float16* __restrict__ dst,
                                     int K, int N) {
  int i = blockIdx.x * blockDim.x + threadIdx.x;
  if (i >= K * N) return;
  int k = i / N, n = i - k * N;
  dst[(long)n * K + k] = (_Float16)src[i];
}

// ---------------------------------------------------------------------------
// WMMA GEMM, f16 operands both contiguous in K:
//   A : element (m,k) at A[m*lda + k]          (f16)
//   Bt: element (k,n) at Bt[n*ldbt + k]        (f16, pre-transposed weight)
//   C[m,n] = dot (+bias[n]) (+addm[m,n]) (relu); stores f32 C and/or f16 C16.
// One wave per 16x16 tile; each operand = two global_load_b128 of 8 halves.
// grid.x = N/16, grid.y = ceil(M/16), block = 32.
// ---------------------------------------------------------------------------
__global__ void __launch_bounds__(32)
wmma_gemm_f16(const _Float16* __restrict__ A, int lda,
              const _Float16* __restrict__ Bt, int ldbt,
              const float* __restrict__ bias,
              const float* __restrict__ addm, int ldadd,
              float* __restrict__ C, _Float16* __restrict__ C16, int ldc,
              int M, int K, int N, int relu) {
  const int lane = threadIdx.x;
  const int m0 = blockIdx.y << 4;
  const int n0 = blockIdx.x << 4;
  const int l15 = lane & 15;
  const int kb  = (lane >> 4) << 3;   // 0 or 8 (lane-pair K offset)

  int arow = m0 + l15;
  if (arow >= M) arow = M - 1;        // clamp; D rows >= M never stored
  const int bcol = n0 + l15;

  const _Float16* ap = A  + (long)arow * lda + kb;
  const _Float16* bp = Bt + (long)bcol * ldbt + kb;

  v8f acc = {};
  const int Kmain = K & ~31;
  for (int k0 = 0; k0 < Kmain; k0 += 32) {
    h8 a0 = *(const h8*)(ap + k0);
    h8 a1 = *(const h8*)(ap + k0 + 16);
    h8 b0 = *(const h8*)(bp + k0);
    h8 b1 = *(const h8*)(bp + k0 + 16);
    v16h a = __builtin_shufflevector(a0, a1, 0, 1, 2, 3, 4, 5, 6, 7,
                                             8, 9, 10, 11, 12, 13, 14, 15);
    v16h b = __builtin_shufflevector(b0, b1, 0, 1, 2, 3, 4, 5, 6, 7,
                                             8, 9, 10, 11, 12, 13, 14, 15);
    acc = __builtin_amdgcn_wmma_f32_16x16x32_f16(false, a, false, b,
                                                 (short)0, acc, false, false);
  }
  if (Kmain < K) {                    // K tail (e.g. K=528), zero-padded
    v16h a, b;
#pragma unroll
    for (int i = 0; i < 8; ++i) {
      int k1 = Kmain + kb + i, k2 = Kmain + 16 + kb + i;
      a[i]     = (k1 < K) ? A [(long)arow * lda + k1] : (_Float16)0.0f;
      a[i + 8] = (k2 < K) ? A [(long)arow * lda + k2] : (_Float16)0.0f;
      b[i]     = (k1 < K) ? Bt[(long)bcol * ldbt + k1] : (_Float16)0.0f;
      b[i + 8] = (k2 < K) ? Bt[(long)bcol * ldbt + k2] : (_Float16)0.0f;
    }
    acc = __builtin_amdgcn_wmma_f32_16x16x32_f16(false, a, false, b,
                                                 (short)0, acc, false, false);
  }

  const int mb = (lane >> 4) << 3;    // D layout: vgpr r -> M = mb + r
#pragma unroll
  for (int r = 0; r < 8; ++r) {
    int m = m0 + mb + r;
    if (m < M) {
      float o = acc[r];
      if (bias) o += bias[bcol];
      if (addm) o += addm[(long)m * ldadd + bcol];
      if (relu) o = fmaxf(o, 0.0f);
      if (C)   C  [(long)m * ldc + bcol] = o;
      if (C16) C16[(long)m * ldc + bcol] = (_Float16)o;
    }
  }
}

// ---------------------------------------------------------------------------
// Edge encoder: ef = relu(edge_attr @ W1 + b1) @ W2 + b2    [E,4]->[E,32]
// ---------------------------------------------------------------------------
__global__ void edge_encoder_kernel(const float* __restrict__ ea,
                                    const float* __restrict__ w1,
                                    const float* __restrict__ b1,
                                    const float* __restrict__ w2,
                                    const float* __restrict__ b2,
                                    float* __restrict__ ef, int E) {
  int e = blockIdx.x * blockDim.x + threadIdx.x;
  if (e >= E) return;
  float a0 = ea[e * 4 + 0], a1 = ea[e * 4 + 1];
  float a2 = ea[e * 4 + 2], a3 = ea[e * 4 + 3];
  float h1[16];
#pragma unroll
  for (int j = 0; j < 16; ++j) {
    float s = b1[j] + a0 * w1[j] + a1 * w1[16 + j] + a2 * w1[32 + j] + a3 * w1[48 + j];
    h1[j] = fmaxf(s, 0.0f);
  }
#pragma unroll
  for (int j = 0; j < 32; ++j) {
    float s = b2[j];
#pragma unroll
    for (int i = 0; i < 16; ++i) s += h1[i] * w2[i * 32 + j];
    ef[(long)e * 32 + j] = s;
  }
}

// ---------------------------------------------------------------------------
// Per-layer attention scratch init: aggv=0, g=0, den=0, mrow=-inf
// ---------------------------------------------------------------------------
__global__ void attn_init_kernel(float* aggv, float* g, float* mrow, float* den, int n) {
  int i = blockIdx.x * blockDim.x + threadIdx.x;
  if (i < n * HIDD) aggv[i] = 0.0f;
  if (i < n * HEADS * EDIM) g[i] = 0.0f;
  if (i < n * HEADS) { mrow[i] = -__builtin_inff(); den[i] = 0.0f; }
}

// ---------------------------------------------------------------------------
// score[e,h] = scale * ( q[tgt,h,:].k[src,h,:] + ef[e,:].qe[tgt,h,:] )
// ---------------------------------------------------------------------------
__global__ void score_kernel(const int* __restrict__ src, const int* __restrict__ tgt,
                             const float* __restrict__ q, const float* __restrict__ k,
                             const float* __restrict__ qe, const float* __restrict__ ef,
                             float* __restrict__ score, float* __restrict__ mrow, int E) {
  int idx = blockIdx.x * blockDim.x + threadIdx.x;
  if (idx >= E * HEADS) return;
  int e = idx >> 3, h = idx & 7;
  int s = src[e], t = tgt[e];
  const float* qp  = q  + (long)t * HIDD + h * CH;
  const float* kp  = k  + (long)s * HIDD + h * CH;
  const float* qep = qe + (long)t * (HEADS * EDIM) + h * EDIM;
  const float* efp = ef + (long)e * EDIM;
  float acc = 0.0f;
#pragma unroll 8
  for (int c = 0; c < CH; ++c) acc += qp[c] * kp[c];
#pragma unroll 8
  for (int j = 0; j < EDIM; ++j) acc += qep[j] * efp[j];
  acc *= 0.125f;                       // 1/sqrt(64)
  score[idx] = acc;
  atomicMaxF(&mrow[t * HEADS + h], acc);
}

// ---------------------------------------------------------------------------
// alpha = exp(score - m[tgt]);  den[tgt] += alpha;  (alpha overwrites score)
// ---------------------------------------------------------------------------
__global__ void alpha_kernel(const int* __restrict__ tgt, float* __restrict__ score,
                             const float* __restrict__ mrow, float* __restrict__ den, int E) {
  int idx = blockIdx.x * blockDim.x + threadIdx.x;
  if (idx >= E * HEADS) return;
  int e = idx >> 3, h = idx & 7;
  int t = tgt[e];
  float al = __expf(score[idx] - mrow[t * HEADS + h]);
  score[idx] = al;
  atomicAdd(&den[t * HEADS + h], al);
}

// ---------------------------------------------------------------------------
// w = alpha/den[tgt];  aggv[tgt,h,:] += w*v[src,h,:];  g[tgt,h,:] += w*ef[e,:]
// ---------------------------------------------------------------------------
__global__ void message_kernel(const int* __restrict__ src, const int* __restrict__ tgt,
                               const float* __restrict__ v, const float* __restrict__ ef,
                               const float* __restrict__ alpha, const float* __restrict__ den,
                               float* __restrict__ aggv, float* __restrict__ g, int E) {
  int idx = blockIdx.x * blockDim.x + threadIdx.x;
  if (idx >= E * HEADS) return;
  int e = idx >> 3, h = idx & 7;
  int s = src[e], t = tgt[e];
  float w = alpha[idx] / den[t * HEADS + h];
  const float* vp  = v  + (long)s * HIDD + h * CH;
  const float* efp = ef + (long)e * EDIM;
  float* av = aggv + (long)t * HIDD + h * CH;
  float* gp = g    + (long)t * (HEADS * EDIM) + h * EDIM;
#pragma unroll 4
  for (int c = 0; c < CH; ++c) atomicAdd(&av[c], w * vp[c]);
#pragma unroll 4
  for (int j = 0; j < EDIM; ++j) atomicAdd(&gp[j], w * efp[j]);
}

// ---------------------------------------------------------------------------
// h = LayerNorm(aggv + skp + h)*gamma + beta  (in-place), also emit f16 copy
// ---------------------------------------------------------------------------
__global__ void __launch_bounds__(128)
fuse_ln_kernel(const float* __restrict__ aggv, const float* __restrict__ skp,
               float* __restrict__ h, _Float16* __restrict__ h16,
               const float* __restrict__ gam, const float* __restrict__ bet) {
  int n = blockIdx.x, t = threadIdx.x;
  __shared__ float sb[128], sb2[128];
  float vals[4];
  float s = 0.0f, s2 = 0.0f;
#pragma unroll
  for (int i = 0; i < 4; ++i) {
    int c = t + i * 128;
    float x = aggv[(long)n * HIDD + c] + skp[(long)n * HIDD + c] + h[(long)n * HIDD + c];
    vals[i] = x; s += x; s2 += x * x;
  }
  sb[t] = s; sb2[t] = s2;
  __syncthreads();
  for (int off = 64; off > 0; off >>= 1) {
    if (t < off) { sb[t] += sb[t + off]; sb2[t] += sb2[t + off]; }
    __syncthreads();
  }
  float mu  = sb[0] * (1.0f / HIDD);
  float var = sb2[0] * (1.0f / HIDD) - mu * mu;
  float inv = rsqrtf(var + 1e-5f);
#pragma unroll
  for (int i = 0; i < 4; ++i) {
    int c = t + i * 128;
    float o = (vals[i] - mu) * inv * gam[c] + bet[c];
    h  [(long)n * HIDD + c] = o;
    h16[(long)n * HIDD + c] = (_Float16)o;
  }
}

// ---------------------------------------------------------------------------
// comb16 = f16 concat(h, dom)  -> [N, 528]   (only consumed as GEMM A)
// ---------------------------------------------------------------------------
__global__ void concat_kernel(const float* __restrict__ h, const float* __restrict__ dom,
                              _Float16* __restrict__ comb16, int Nn) {
  int i = blockIdx.x * blockDim.x + threadIdx.x;
  if (i >= Nn * 528) return;
  int n = i / 528, c = i - n * 528;
  float o = (c < HIDD) ? h[(long)n * HIDD + c] : dom[(long)n * 16 + (c - HIDD)];
  comb16[i] = (_Float16)o;
}

// ---------------------------------------------------------------------------
static void gemm(hipStream_t st, const _Float16* A, int lda,
                 const _Float16* Bt, int ldbt,
                 const float* bias, const float* addm, int ldadd,
                 float* C, _Float16* C16, int ldc,
                 int M, int K, int N, int relu) {
  dim3 grid(N / 16, (M + 15) / 16);
  wmma_gemm_f16<<<grid, dim3(32), 0, st>>>(A, lda, Bt, ldbt, bias, addm, ldadd,
                                           C, C16, ldc, M, K, N, relu);
}

static void tcvt(hipStream_t st, const float* src, _Float16* dst, int K, int N) {
  transpose_cvt_kernel<<<(K * N + 255) / 256, 256, 0, st>>>(src, dst, K, N);
}
static void cvt(hipStream_t st, const float* src, _Float16* dst, int n) {
  cvt_kernel<<<(n + 255) / 256, 256, 0, st>>>(src, dst, n);
}

extern "C" void kernel_launch(void* const* d_in, const int* in_sizes, int n_in,
                              void* d_out, int out_size, void* d_ws, size_t ws_size,
                              hipStream_t stream) {
  (void)n_in; (void)out_size; (void)ws_size;
  const float* x      = (const float*)d_in[0];
  const int*   ei     = (const int*)  d_in[1];
  const float* eattr  = (const float*)d_in[2];
  const float* emb_w  = (const float*)d_in[3];
  const float* emb_b  = (const float*)d_in[4];
  const float* pe     = (const float*)d_in[5];
  const float* ee_w1  = (const float*)d_in[6];
  const float* ee_b1  = (const float*)d_in[7];
  const float* ee_w2  = (const float*)d_in[8];
  const float* ee_b2  = (const float*)d_in[9];
  const float* q_w    = (const float*)d_in[10];
  const float* q_b    = (const float*)d_in[11];
  const float* k_w    = (const float*)d_in[12];
  const float* k_b    = (const float*)d_in[13];
  const float* v_w    = (const float*)d_in[14];
  const float* v_b    = (const float*)d_in[15];
  const float* e_w    = (const float*)d_in[16];
  const float* skip_w = (const float*)d_in[17];
  const float* skip_b = (const float*)d_in[18];
  const float* ln_g   = (const float*)d_in[19];
  const float* ln_b   = (const float*)d_in[20];
  const float* dom_w  = (const float*)d_in[21];
  const float* dom_b  = (const float*)d_in[22];
  const float* cls_w1 = (const float*)d_in[23];
  const float* cls_b1 = (const float*)d_in[24];
  const float* cls_w2 = (const float*)d_in[25];
  const float* cls_b2 = (const float*)d_in[26];

  const int Nn = in_sizes[0] / 256;     // 1000
  const int E  = in_sizes[1] / 2;       // 128000
  const int* src = ei;
  const int* tgt = ei + E;

  // ---------------- f32 workspace ----------------
  float* ws   = (float*)d_ws;
  float* h    = ws;  ws += (size_t)Nn * HIDD;
  float* ef   = ws;  ws += (size_t)E * EDIM;
  float* q    = ws;  ws += (size_t)Nn * HIDD;
  float* k    = ws;  ws += (size_t)Nn * HIDD;
  float* v    = ws;  ws += (size_t)Nn * HIDD;
  float* skp  = ws;  ws += (size_t)Nn * HIDD;
  float* qe   = ws;  ws += (size_t)Nn * HEADS * EDIM;
  float* sc   = ws;  ws += (size_t)E * HEADS;
  float* mrow = ws;  ws += (size_t)Nn * HEADS;
  float* den  = ws;  ws += (size_t)Nn * HEADS;
  float* aggv = ws;  ws += (size_t)Nn * HIDD;
  float* gbuf = ws;  ws += (size_t)Nn * HEADS * EDIM;
  float* dom  = ws;  ws += (size_t)Nn * 16;

  // ---------------- f16 workspace (16B aligned) ----------------
  uintptr_t p16 = ((uintptr_t)ws + 15) & ~(uintptr_t)15;
  _Float16* f = (_Float16*)p16;
  _Float16* x16    = f;  f += (size_t)Nn * 256;
  _Float16* h16    = f;  f += (size_t)Nn * HIDD;
  _Float16* q16    = f;  f += (size_t)Nn * HIDD;
  _Float16* gbuf16 = f;  f += (size_t)Nn * HEADS * EDIM;
  _Float16* comb16 = f;  f += (size_t)Nn * 528;
  _Float16* hidb16 = f;  f += (size_t)Nn * 256;
  _Float16* embWt  = f;  f += (size_t)HIDD * 256;        // [512,256]
  _Float16* qWt    = f;  f += (size_t)HIDD * HIDD;       // per-layer reuse
  _Float16* kWt    = f;  f += (size_t)HIDD * HIDD;
  _Float16* vWt    = f;  f += (size_t)HIDD * HIDD;
  _Float16* sWt    = f;  f += (size_t)HIDD * HIDD;
  _Float16* ew16   = f;  f += (size_t)EDIM * HIDD;       // e_w as-is [32,512]
  _Float16* ewT16  = f;  f += (size_t)HIDD * EDIM;       // e_w^T   [512,32]
  _Float16* domWt  = f;  f += (size_t)16 * HIDD;
  _Float16* c1Wt   = f;  f += (size_t)256 * 528;
  _Float16* c2Wt   = f;  f += (size_t)16 * 256;

  // ---------------- prep: weight transpose-convert + activations ----------
  cvt (stream, x, x16, Nn * 256);
  tcvt(stream, emb_w, embWt, 256, HIDD);
  tcvt(stream, dom_w, domWt, HIDD, 16);
  tcvt(stream, cls_w1, c1Wt, 528, 256);
  tcvt(stream, cls_w2, c2Wt, 256, 16);

  // 1) edge encoder
  edge_encoder_kernel<<<(E + 255) / 256, 256, 0, stream>>>(eattr, ee_w1, ee_b1,
                                                           ee_w2, ee_b2, ef, E);
  // 2) embed: h = x @ embed_w + b + pe   (dual-store f32 + f16)
  gemm(stream, x16, 256, embWt, 256, emb_b, pe, HIDD, h, h16, HIDD, Nn, 256, HIDD, 0);

  const int EH = E * HEADS;
  for (int l = 0; l < 6; ++l) {
    const float* qwl = q_w    + (size_t)l * HIDD * HIDD;
    const float* kwl = k_w    + (size_t)l * HIDD * HIDD;
    const float* vwl = v_w    + (size_t)l * HIDD * HIDD;
    const float* swl = skip_w + (size_t)l * HIDD * HIDD;
    const float* ewl = e_w    + (size_t)l * EDIM * HIDD;

    // per-layer weight prep (f16, transposed for contiguous-K B loads)
    tcvt(stream, qwl, qWt, HIDD, HIDD);
    tcvt(stream, kwl, kWt, HIDD, HIDD);
    tcvt(stream, vwl, vWt, HIDD, HIDD);
    tcvt(stream, swl, sWt, HIDD, HIDD);
    cvt (stream, ewl, ew16, EDIM * HIDD);
    tcvt(stream, ewl, ewT16, EDIM, HIDD);

    // q, k, v, skip projections (WMMA)
    gemm(stream, h16, HIDD, qWt, HIDD, q_b + l * HIDD, nullptr, 0, q, q16, HIDD, Nn, HIDD, HIDD, 0);
    gemm(stream, h16, HIDD, kWt, HIDD, k_b + l * HIDD, nullptr, 0, k, nullptr, HIDD, Nn, HIDD, HIDD, 0);
    gemm(stream, h16, HIDD, vWt, HIDD, v_b + l * HIDD, nullptr, 0, v, nullptr, HIDD, Nn, HIDD, HIDD, 0);
    gemm(stream, h16, HIDD, sWt, HIDD, skip_b + l * HIDD, nullptr, 0, skp, nullptr, HIDD, Nn, HIDD, HIDD, 0);

    // qe[n,h,j] = sum_c q[n,h,c] * e_w[j, h*64+c]   (8 per-head WMMA GEMMs)
    for (int hh = 0; hh < HEADS; ++hh)
      gemm(stream, q16 + hh * CH, HIDD, ew16 + hh * CH, HIDD, nullptr, nullptr, 0,
           qe + hh * EDIM, nullptr, HEADS * EDIM, Nn, CH, EDIM, 0);

    // attention scratch init
    attn_init_kernel<<<(Nn * HIDD + 255) / 256, 256, 0, stream>>>(aggv, gbuf, mrow, den, Nn);
    // scores + segment max
    score_kernel<<<(EH + 255) / 256, 256, 0, stream>>>(src, tgt, q, k, qe, ef, sc, mrow, E);
    // exp + segment sum
    alpha_kernel<<<(EH + 255) / 256, 256, 0, stream>>>(tgt, sc, mrow, den, E);
    // normalized messages: v-part and ef-part scatter
    message_kernel<<<(EH + 255) / 256, 256, 0, stream>>>(src, tgt, v, ef, sc, den, aggv, gbuf, E);

    // edge contribution: aggv[:,h*64:] += g[:,h,:] @ e_w[:, h*64:]  (WMMA, in-place)
    cvt(stream, gbuf, gbuf16, Nn * HEADS * EDIM);
    for (int hh = 0; hh < HEADS; ++hh)
      gemm(stream, gbuf16 + hh * EDIM, HEADS * EDIM, ewT16 + hh * CH * EDIM, EDIM,
           nullptr, aggv + hh * CH, HIDD, aggv + hh * CH, nullptr, HIDD, Nn, EDIM, CH, 0);

    // residual + skip + layer norm -> h, h16 (in place)
    fuse_ln_kernel<<<Nn, 128, 0, stream>>>(aggv, skp, h, h16, ln_g + l * HIDD, ln_b + l * HIDD);
  }

  // head: dom, concat, classifier
  gemm(stream, h16, HIDD, domWt, HIDD, dom_b, nullptr, 0, dom, nullptr, 16, Nn, HIDD, 16, 0);
  concat_kernel<<<(Nn * 528 + 255) / 256, 256, 0, stream>>>(h, dom, comb16, Nn);
  gemm(stream, comb16, 528, c1Wt, 528, cls_b1, nullptr, 0, nullptr, hidb16, 256, Nn, 528, 256, 1);
  gemm(stream, hidb16, 256, c2Wt, 256, cls_b2, nullptr, 0, (float*)d_out, nullptr, 16, Nn, 256, 16, 0);
}
